// TriHardLoss_6657199309074
// MI455X (gfx1250) — matrix operand order repaired and checked by reading, not compiled, not run on hardware
//
#include <hip/hip_runtime.h>

// ---------------------------------------------------------------------------
// TriHard loss for MI455X (gfx1250):
//   1) one-shot f32 -> bf16 hi/lo split (x = hi + lo; fp32-grade GEMM accuracy)
//   2) per-row norms + keyed max/min init
//   3) symmetric-half tiled Gram via v_wmma_f32_16x16x32_bf16:
//        G ~= hi*hi' + hi*lo' + lo*hi'   (3 bf16 WMMAs per K-slab; ~2.7x
//        cheaper than the native 16x16x4 f32 WMMA path at ~2^-17 rel. error)
//      Tensor-Data-Mover double-buffered staging, fused row+column mining
//   4) scalar loss reduction
// ---------------------------------------------------------------------------

typedef __bf16 bf16_t;
typedef __attribute__((ext_vector_type(16))) __bf16 v16bf;
typedef __attribute__((ext_vector_type(8)))  __bf16 v8bf;
typedef __attribute__((ext_vector_type(4)))  __bf16 v4bf;
typedef __attribute__((ext_vector_type(8)))  float  v8f;
typedef __attribute__((ext_vector_type(4)))  float  v4f;
typedef __attribute__((ext_vector_type(4)))  unsigned int v4u;
typedef __attribute__((ext_vector_type(8)))  unsigned int v8u;

#define MARGIN_F   0.3f
#define NEG_INF_F (-1e30f)
#define POS_INF_F ( 1e30f)

#define BM 128   // tile rows (M)
#define BN 128   // tile cols (N)
#define BK 32    // K step (matches v_wmma_f32_16x16x32_bf16)
#define LDA 40   // LDS row stride in bf16 elems (80 B); TDM pad: 16 dw + 4 dw

// Order-preserving float <-> u32 key (u32 compare == float compare).
__device__ __forceinline__ unsigned int f32_key(float f) {
    unsigned int u = __float_as_uint(f);
    return (u & 0x80000000u) ? ~u : (u | 0x80000000u);
}
__device__ __forceinline__ float key_f32(unsigned int k) {
    unsigned int u = (k & 0x80000000u) ? (k ^ 0x80000000u) : ~k;
    return __uint_as_float(u);
}

// ---------------------------------------------------------------------------
// TDM: DMA one 128x32 bf16 tile (row stride K elems) Global -> LDS, with
// 4-DWORD padding after every 16 DWORDs (=> 80 B LDS row stride == LDA).
// D# per CDNA5 ISA; 2-D => groups 2/3 NULL.
// ---------------------------------------------------------------------------
__device__ __forceinline__ void tdm_load_tile(unsigned lds_addr,
                                              const bf16_t* gsrc,
                                              unsigned K, unsigned N) {
    const unsigned long long ga = (unsigned long long)(size_t)gsrc;
    v4u g0;
    g0[0] = 1u;                                   // count=1, user descriptor
    g0[1] = lds_addr;                             // LDS byte address
    g0[2] = (unsigned)(ga & 0xFFFFFFFFull);       // global_addr[31:0]
    g0[3] = (unsigned)((ga >> 32) & 0x01FFFFFFull)
          | 0x80000000u;                          // global_addr[56:32] | type=2
    v8u g1;
    g1[0] = (1u << 16)      // data_size = 2 bytes
          | (1u << 20)      // pad_enable
          | (3u << 22)      // pad_interval: 16 DWORDs
          | (3u << 25);     // pad_amount:   4 DWORDs
    g1[1] = (K & 0xFFFFu) << 16;                  // tensor_dim0[15:0]
    g1[2] = (K >> 16) | ((N & 0xFFFFu) << 16);    // dim0[31:16] | dim1[15:0]
    g1[3] = (N >> 16) | ((unsigned)BK << 16);     // dim1[31:16] | tile_dim0=32
    g1[4] = (unsigned)BM;                         // tile_dim1=128, tile_dim2=0
    g1[5] = K;                                    // tensor_dim0_stride[31:0]
    g1[6] = 0u;
    g1[7] = 0u;
    asm volatile("tensor_load_to_lds %0, %1" :: "s"(g0), "s"(g1) : "memory");
}

// Fragment loaders following the ISA 16-bit A(16x32) / B(32x16) lane layouts.
__device__ __forceinline__ v16bf load_afrag(const bf16_t* base) {
    v8bf a0 = *(const v8bf*)base;          // K = half*8 .. +7
    v8bf a1 = *(const v8bf*)(base + 16);   // K = 16+half*8 .. +7
    return __builtin_shufflevector(
        a0, a1, 0, 1, 2, 3, 4, 5, 6, 7, 8, 9, 10, 11, 12, 13, 14, 15);
}
__device__ __forceinline__ v16bf load_bfrag(const bf16_t* base) {
    v8bf b0 = *(const v8bf*)base;          // K = half*8 .. +7
    v8bf b1 = *(const v8bf*)(base + 16);   // K = 16+half*8 .. +7
    return __builtin_shufflevector(
        b0, b1, 0, 1, 2, 3, 4, 5, 6, 7, 8, 9, 10, 11, 12, 13, 14, 15);
}

// ---------------------------------------------------------------------------
// Kernel 0: one-shot f32 -> (hi, lo) bf16 split, 8 elems/thread
// ---------------------------------------------------------------------------
__global__ __launch_bounds__(256)
void trihard_split_bf16(const float* __restrict__ x,
                        bf16_t* __restrict__ xhi,
                        bf16_t* __restrict__ xlo) {
    const size_t idx = ((size_t)blockIdx.x * 256 + threadIdx.x) * 8;
    v4f a = *(const v4f*)(x + idx);
    v4f b = *(const v4f*)(x + idx + 4);
    v4bf ha = __builtin_convertvector(a, v4bf);
    v4bf hb = __builtin_convertvector(b, v4bf);
    *(v8bf*)(xhi + idx) =
        __builtin_shufflevector(ha, hb, 0, 1, 2, 3, 4, 5, 6, 7);
    v4f ra = a - __builtin_convertvector(ha, v4f);
    v4f rb = b - __builtin_convertvector(hb, v4f);
    v4bf la = __builtin_convertvector(ra, v4bf);
    v4bf lb = __builtin_convertvector(rb, v4bf);
    *(v8bf*)(xlo + idx) =
        __builtin_shufflevector(la, lb, 0, 1, 2, 3, 4, 5, 6, 7);
}

// ---------------------------------------------------------------------------
// Kernel 1: per-row squared norms (f32 source) + keyed accumulator init
// ---------------------------------------------------------------------------
__global__ __launch_bounds__(256)
void trihard_rownorm_init(const float* __restrict__ x, int K,
                          float* __restrict__ sq,
                          unsigned int* __restrict__ apk,
                          unsigned int* __restrict__ ank) {
    const int row = blockIdx.x;
    const float* r = x + (size_t)row * K;
    float s = 0.f;
    for (int k = threadIdx.x; k < K; k += 256) {
        float v = r[k];
        s += v * v;
    }
    __shared__ float red[256];
    red[threadIdx.x] = s;
    __syncthreads();
    for (int off = 128; off > 0; off >>= 1) {
        if (threadIdx.x < off) red[threadIdx.x] += red[threadIdx.x + off];
        __syncthreads();
    }
    if (threadIdx.x == 0) {
        sq[row]  = red[0];
        apk[row] = 0u;           // key(-inf)
        ank[row] = 0xFFFFFFFFu;  // key(+inf)
    }
}

// ---------------------------------------------------------------------------
// Kernel 2: upper-triangular 128x128 Gram tiles with split-bf16 WMMA.
// TDM double-buffers 4 tiles/K-step (A/B x hi/lo); waves run
// ds_load_b128 -> 3x v_wmma per N-subtile. Fused row+column hard mining.
// ---------------------------------------------------------------------------
__global__ __launch_bounds__(256)
void trihard_gemm_mine(const bf16_t* __restrict__ xhi,
                       const bf16_t* __restrict__ xlo,
                       const int* __restrict__ lab,
                       const float* __restrict__ sq,
                       unsigned int* __restrict__ apk,
                       unsigned int* __restrict__ ank,
                       int K, int N) {
    if (blockIdx.x < blockIdx.y) return;   // symmetric half only (uniform exit)

    __shared__ bf16_t lds_a[2][2][BM * LDA];   // [buffer][hi/lo]
    __shared__ bf16_t lds_b[2][2][BN * LDA];
    __shared__ float  s_sqm[BM];
    __shared__ float  s_sqn[BN];
    __shared__ int    s_lm[BM];
    __shared__ int    s_ln[BN];

    const int tid   = threadIdx.x;
    const int wave  = tid >> 5;
    const int lane  = tid & 31;
    const int half  = lane >> 4;
    const int l16   = lane & 15;
    const int m0    = blockIdx.y * BM;
    const int n0    = blockIdx.x * BN;
    const int mbase = wave * 16;           // each wave owns a 16-row strip

    if (tid < BM) {
        s_sqm[tid] = sq[m0 + tid];
        s_lm[tid]  = lab[m0 + tid];
    } else {
        int t = tid - BM;
        s_sqn[t] = sq[n0 + t];
        s_ln[t]  = lab[n0 + t];
    }

    const bf16_t* gaH = xhi + (size_t)m0 * K;
    const bf16_t* gaL = xlo + (size_t)m0 * K;
    const bf16_t* gbH = xhi + (size_t)n0 * K;
    const bf16_t* gbL = xlo + (size_t)n0 * K;

    v8f acc[8] = {};
    const int kIters = K / BK;

    if (wave == 0) {                              // DMA tile 0 into buffer 0
        tdm_load_tile((unsigned)(size_t)&lds_a[0][0][0], gaH, (unsigned)K, (unsigned)N);
        tdm_load_tile((unsigned)(size_t)&lds_a[0][1][0], gaL, (unsigned)K, (unsigned)N);
        tdm_load_tile((unsigned)(size_t)&lds_b[0][0][0], gbH, (unsigned)K, (unsigned)N);
        tdm_load_tile((unsigned)(size_t)&lds_b[0][1][0], gbL, (unsigned)K, (unsigned)N);
    }

    int buf = 0;
    for (int kt = 0; kt < kIters; ++kt) {
        const bool more = (kt + 1) < kIters;
        if (wave == 0) {
            if (more) {                           // DMA tile kt+1 -> other buf
                const int k1 = (kt + 1) * BK;
                const int nb2 = buf ^ 1;
                tdm_load_tile((unsigned)(size_t)&lds_a[nb2][0][0], gaH + k1, (unsigned)K, (unsigned)N);
                tdm_load_tile((unsigned)(size_t)&lds_a[nb2][1][0], gaL + k1, (unsigned)K, (unsigned)N);
                tdm_load_tile((unsigned)(size_t)&lds_b[nb2][0][0], gbH + k1, (unsigned)K, (unsigned)N);
                tdm_load_tile((unsigned)(size_t)&lds_b[nb2][1][0], gbL + k1, (unsigned)K, (unsigned)N);
                __builtin_amdgcn_s_wait_tensorcnt(4);  // tile kt's 4 DMAs done
            } else {
                __builtin_amdgcn_s_wait_tensorcnt(0);
            }
        }
        __syncthreads();                          // publish buffer `buf`

        const int aoff = (mbase + l16) * LDA + half * 8;
        v16bf aH = load_afrag(&lds_a[buf][0][aoff]);
        v16bf aL = load_afrag(&lds_a[buf][1][aoff]);

        #pragma unroll
        for (int nb = 0; nb < 8; ++nb) {
            const int boff = (nb * 16 + l16) * LDA + half * 8;
            v16bf bH = load_bfrag(&lds_b[buf][0][boff]);
            v16bf bL = load_bfrag(&lds_b[buf][1][boff]);
            // G += hi*hi' + hi*lo' + lo*hi'  (lo*lo' ~ 2^-18, dropped)
            acc[nb] = __builtin_amdgcn_wmma_f32_16x16x32_bf16(
                false, aH, false, bH, (short)0, acc[nb], false, false);
            acc[nb] = __builtin_amdgcn_wmma_f32_16x16x32_bf16(
                false, aH, false, bL, (short)0, acc[nb], false, false);
            acc[nb] = __builtin_amdgcn_wmma_f32_16x16x32_bf16(
                false, aL, false, bH, (short)0, acc[nb], false, false);
        }

        __syncthreads();                          // done reading `buf`
        buf ^= 1;
    }

    // ---- Epilogue: dist = sq_m + sq_n - 2G, masked row & column mining ----
    float sm[8];
    int   lm[8];
    #pragma unroll
    for (int v = 0; v < 8; ++v) {
        const int ml = mbase + half * 8 + v;
        sm[v] = s_sqm[ml];
        lm[v] = s_lm[ml];
    }

    float apr[8], anr[8];
    #pragma unroll
    for (int v = 0; v < 8; ++v) { apr[v] = NEG_INF_F; anr[v] = POS_INF_F; }

    #pragma unroll
    for (int nb = 0; nb < 8; ++nb) {
        const int   nl = nb * 16 + l16;
        const float sn = s_sqn[nl];
        const int   ln = s_ln[nl];
        float apc = NEG_INF_F, anc = POS_INF_F;
        #pragma unroll
        for (int v = 0; v < 8; ++v) {
            const float d    = sm[v] + sn - 2.0f * acc[nb][v];
            const bool  same = (lm[v] == ln);
            apr[v] = same ? fmaxf(apr[v], d) : apr[v];
            anr[v] = same ? anr[v] : fminf(anr[v], d);
            apc    = same ? fmaxf(apc, d) : apc;
            anc    = same ? anc : fminf(anc, d);
        }
        // Column nl: combine the two lane-halves (M=0..7 with M=8..15).
        apc = fmaxf(apc, __shfl_xor(apc, 16, 32));
        anc = fminf(anc, __shfl_xor(anc, 16, 32));
        if (half == 0) {
            const int gcol = n0 + nl;
            atomicMax(&apk[gcol], f32_key(apc));
            atomicMin(&ank[gcol], f32_key(anc));
        }
    }

    // Rows: reduce across the 16-lane half-group sharing each row.
    #pragma unroll
    for (int v = 0; v < 8; ++v) {
        float ap = apr[v], an = anr[v];
        #pragma unroll
        for (int off = 1; off < 16; off <<= 1) {
            ap = fmaxf(ap, __shfl_xor(ap, off, 32));
            an = fminf(an, __shfl_xor(an, off, 32));
        }
        if (l16 == 0) {
            const int grow = m0 + mbase + half * 8 + v;
            atomicMax(&apk[grow], f32_key(ap));
            atomicMin(&ank[grow], f32_key(an));
        }
    }
}

// ---------------------------------------------------------------------------
// Kernel 3: loss = mean(relu(ap - an + margin))
// ---------------------------------------------------------------------------
__global__ __launch_bounds__(256)
void trihard_loss_reduce(const unsigned int* __restrict__ apk,
                         const unsigned int* __restrict__ ank,
                         int N, float* __restrict__ out) {
    __shared__ float red[256];
    float s = 0.f;
    for (int r = threadIdx.x; r < N; r += 256) {
        float ap = key_f32(apk[r]);
        float an = key_f32(ank[r]);
        s += fmaxf(ap - an + MARGIN_F, 0.f);
    }
    red[threadIdx.x] = s;
    __syncthreads();
    for (int off = 128; off > 0; off >>= 1) {
        if (threadIdx.x < off) red[threadIdx.x] += red[threadIdx.x + off];
        __syncthreads();
    }
    if (threadIdx.x == 0) out[0] = red[0] / (float)N;
}

// ---------------------------------------------------------------------------
extern "C" void kernel_launch(void* const* d_in, const int* in_sizes, int n_in,
                              void* d_out, int out_size, void* d_ws, size_t ws_size,
                              hipStream_t stream) {
    (void)n_in; (void)out_size; (void)ws_size;
    const float* x   = (const float*)d_in[0];
    const int*   lab = (const int*)d_in[1];

    const int N = in_sizes[1];          // 4096 rows
    const int K = in_sizes[0] / N;      // 2048 features

    float*        sq  = (float*)d_ws;
    unsigned int* apk = (unsigned int*)((char*)d_ws + (size_t)N * 4);
    unsigned int* ank = (unsigned int*)((char*)d_ws + (size_t)N * 8);
    char*         mb  = (char*)d_ws + (((size_t)N * 12 + 255) & ~(size_t)255);
    bf16_t*       xhi = (bf16_t*)mb;
    bf16_t*       xlo = (bf16_t*)(mb + (size_t)N * K * sizeof(bf16_t));

    const size_t total = (size_t)N * K;
    trihard_split_bf16<<<(unsigned)(total / (256 * 8)), 256, 0, stream>>>(x, xhi, xlo);

    trihard_rownorm_init<<<N, 256, 0, stream>>>(x, K, sq, apk, ank);

    dim3 grid(N / BN, N / BM);
    trihard_gemm_mine<<<grid, 256, 0, stream>>>(xhi, xlo, lab, sq, apk, ank, K, N);

    trihard_loss_reduce<<<1, 256, 0, stream>>>(apk, ank, N, (float*)d_out);
}
